// Decoder_89300960018551
// MI455X (gfx1250) — matrix operand order repaired and checked by reading, not compiled, not run on hardware
//
#include <hip/hip_runtime.h>
#include <hip/hip_bf16.h>

#define B_    8
#define IMG_  512
#define S_    1024
#define D_    768
#define H_    12
#define DH_   64
#define L_    6
#define DFF_  2048
#define OUT_  768
#define BS_   (B_ * S_)

typedef __attribute__((ext_vector_type(16))) __bf16 v16bf;
typedef __attribute__((ext_vector_type(8)))  float  v8f;

// ---------------------------------------------------------------------------
// fp32 -> bf16 (round-to-nearest-even) helpers
// ---------------------------------------------------------------------------
__device__ __forceinline__ unsigned short bf16_bits(float f) {
  union { float f; unsigned u; } v; v.f = f;
  unsigned u = v.u + 0x7FFFu + ((v.u >> 16) & 1u);
  return (unsigned short)(u >> 16);
}
__device__ __forceinline__ unsigned pack_bf16(float a, float b) {
  union { float f; unsigned u; } x, y; x.f = a; y.f = b;
  unsigned ux = x.u + 0x7FFFu + ((x.u >> 16) & 1u);
  unsigned uy = y.u + 0x7FFFu + ((y.u >> 16) & 1u);
  return (ux >> 16) | (uy & 0xFFFF0000u);
}

// ---------------------------------------------------------------------------
// CDNA5 async global -> LDS copy (16 B per lane), tracked by ASYNCcnt.
// ---------------------------------------------------------------------------
__device__ __forceinline__ void async_copy_b128(unsigned lds_off, const unsigned* gsrc) {
  asm volatile("global_load_async_to_lds_b128 %0, %1, off"
               :: "v"(lds_off), "v"(gsrc)
               : "memory");
}
__device__ __forceinline__ void wait_async0() {
  asm volatile("s_wait_asynccnt 0" ::: "memory");
}

// ---------------------------------------------------------------------------
// x = encoded_patches + sinusoidal PE
// ---------------------------------------------------------------------------
__global__ __launch_bounds__(256) void add_pe_kernel(const float* __restrict__ enc,
                                                     float* __restrict__ x) {
  size_t idx = (size_t)blockIdx.x * 256 + threadIdx.x;   // over BS_*D_
  int d = (int)(idx % D_);
  int s = (int)((idx / D_) % S_);
  float freq  = __expf(-9.2103403719761836f * (float)(2 * (d >> 1)) * (1.0f / (float)D_));
  float angle = (float)s * freq;
  float pe    = (d & 1) ? __cosf(angle) : __sinf(angle);
  x[idx] = enc[idx] + pe;
}

// ---------------------------------------------------------------------------
// One-time fp32 -> bf16 fragment-order packers.
// Output layout: [row16_tile][k32_tile][32 lanes][8 dwords]  (1 KB per frag)
//   pack_a: WMMA A-matrix striping (row in lane, K split 8/8 per half-wave)
//   pack_b: WMMA B-matrix striping (K split 8/8 per half-wave, N in lane)
// One block per source row; threads stride over K/2 bf16-pairs.
// ---------------------------------------------------------------------------
__global__ __launch_bounds__(256) void pack_a_kernel(const float* __restrict__ X,
                                                     unsigned* __restrict__ P, int K) {
  const int r  = blockIdx.x;
  const int Kt = K >> 5;
  const int np = K >> 1;
  const float* src = X + (size_t)r * K;
  unsigned* dtile = P + (size_t)(r >> 4) * Kt * 256;
  for (int kp = threadIdx.x; kp < np; kp += 256) {
    const float2 vv = *(const float2*)(src + 2 * kp);
    unsigned pk = pack_bf16(vv.x, vv.y);
    int jpp = kp & 15;
    int tl  = (r & 15) + 16 * ((jpp >> 2) & 1);
    int ep  = (jpp & 3) + 4 * (jpp >> 3);
    dtile[(size_t)(kp >> 4) * 256 + tl * 8 + ep] = pk;
  }
}

__global__ __launch_bounds__(256) void pack_b_kernel(const float* __restrict__ W,
                                                     unsigned* __restrict__ P, int K) {
  const int n  = blockIdx.x;
  const int Kt = K >> 5;
  const int np = K >> 1;
  const float* src = W + (size_t)n * K;
  unsigned* dtile = P + (size_t)(n >> 4) * Kt * 256;
  for (int kp = threadIdx.x; kp < np; kp += 256) {
    const float2 vv = *(const float2*)(src + 2 * kp);
    unsigned pk = pack_bf16(vv.x, vv.y);
    int jpp = kp & 15;
    int tl  = (n & 15) + 16 * ((jpp >> 2) & 1);
    int ep  = (jpp & 3) + 4 * (jpp >> 3);
    dtile[(size_t)(kp >> 4) * 256 + tl * 8 + ep] = pk;
  }
}

// ---------------------------------------------------------------------------
// GEMM on pre-packed bf16 fragments:
//   out[m, n] = sum_k A[m, k] * W[n, k] + bias[n]   (optionally ReLU)
// PA: A packed [M/16][K/32][256]dw, PB: W packed [N/16][K/32][256]dw.
// 256 threads (8 waves), 128x128 tile, K-step 64, double-buffered LDS filled
// with GLOBAL_LOAD_ASYNC_TO_LDS_B128 (ASYNCcnt) overlapped with WMMA compute.
// ---------------------------------------------------------------------------
template<int RELU>
__global__ __launch_bounds__(256) void gemm_kernel(
    const unsigned* __restrict__ PA, const unsigned* __restrict__ PB,
    const float* __restrict__ bias, float* __restrict__ out,
    int K, int ldo)
{
  __shared__ unsigned a_lds[2][16 * 256];   // 2 x [8 m-frags][2 k-frags][256 dw]
  __shared__ unsigned b_lds[2][16 * 256];   // 2 x [8 n-frags][2 k-frags][256 dw]

  const int tid    = threadIdx.x;
  const int lane   = tid & 31;
  const int wid    = tid >> 5;
  const int wave_m = wid >> 2;          // 0..1
  const int wave_n = wid & 3;           // 0..3
  const int Kt     = K >> 5;
  const int mtile0 = blockIdx.y * 8;
  const int ntile0 = blockIdx.x * 8;

  // per-thread chunk geometry: 4 x 16B chunks for A and for B per tile
  auto issue_tile = [&](int kt, int buf) {
    #pragma unroll
    for (int i = 0; i < 4; ++i) {
      int c      = tid + i * 256;       // 16-byte chunk id, 0..1023
      int frag   = c >> 6;              // (m|n)frag*2 + kf
      int within = (c & 63) * 4;        // dword offset inside 1 KB frag
      async_copy_b128((unsigned)(unsigned long long)&a_lds[buf][frag * 256 + within],
                      PA + ((size_t)(mtile0 + (frag >> 1)) * Kt + kt + (frag & 1)) * 256 + within);
      async_copy_b128((unsigned)(unsigned long long)&b_lds[buf][frag * 256 + within],
                      PB + ((size_t)(ntile0 + (frag >> 1)) * Kt + kt + (frag & 1)) * 256 + within);
    }
  };

  v8f vz = {};
  v8f acc[4][2];
  #pragma unroll
  for (int mi = 0; mi < 4; ++mi)
    #pragma unroll
    for (int ni = 0; ni < 2; ++ni) acc[mi][ni] = vz;

  issue_tile(0, 0);
  wait_async0();
  __syncthreads();

  for (int kt = 0; kt < Kt; kt += 2) {
    const int cur = (kt >> 1) & 1;
    if (kt + 2 < Kt) issue_tile(kt + 2, cur ^ 1);   // prefetch next tile

    #pragma unroll
    for (int kf = 0; kf < 2; ++kf) {
      v16bf bfr[2];
      #pragma unroll
      for (int ni = 0; ni < 2; ++ni)
        bfr[ni] = *(const v16bf*)&b_lds[cur][(((wave_n * 2 + ni) * 2 + kf) * 256) + lane * 8];
      #pragma unroll
      for (int mi = 0; mi < 4; ++mi) {
        v16bf af = *(const v16bf*)&a_lds[cur][(((wave_m * 4 + mi) * 2 + kf) * 256) + lane * 8];
        #pragma unroll
        for (int ni = 0; ni < 2; ++ni)
          acc[mi][ni] = __builtin_amdgcn_wmma_f32_16x16x32_bf16(
              false, af, false, bfr[ni], (short)0, acc[mi][ni], false, false);
      }
    }
    wait_async0();       // prefetched tile landed in LDS
    __syncthreads();     // all waves done reading current tile; publish next
  }

  const int nlo = lane & 15;
  const int hh  = lane >> 4;
  const int m_base = blockIdx.y * 128;
  const int n_base = blockIdx.x * 128;
  #pragma unroll
  for (int ni = 0; ni < 2; ++ni) {
    int n = n_base + (wave_n * 2 + ni) * 16 + nlo;
    float bv = bias[n];
    #pragma unroll
    for (int mi = 0; mi < 4; ++mi) {
      #pragma unroll
      for (int v = 0; v < 8; ++v) {
        int m = m_base + (wave_m * 4 + mi) * 16 + v + 8 * hh;
        float r = acc[mi][ni][v] + bv;
        if (RELU) r = fmaxf(r, 0.0f);
        out[(size_t)m * ldo + n] = r;
      }
    }
  }
}

// ---------------------------------------------------------------------------
// Flash attention over packed qkv buffer [BS, 3*D] (q | k | v per row).
// Block = 128 queries (8 waves x 16), streams 32-key chunks; QK^T and P*V
// both via bf16 WMMA; 1/sqrt(DH) folded into Q at load.
// ---------------------------------------------------------------------------
template<bool CAUSAL>
__global__ __launch_bounds__(256) void attn_kernel(const float* __restrict__ qkv,
                                                   float* __restrict__ ctx)
{
  __shared__ unsigned       q_lds[8 * 2 * 256];  // 8 q-frags x 2 dh-frags
  __shared__ unsigned       k_lds[2 * 2 * 256];  // 2 key n-tiles x 2 dh-frags
  __shared__ unsigned short v_lds[4 * 512];      // 4 dh n-tiles (B frags)
  __shared__ unsigned short p_lds[8 * 512];      // per-wave P as A-frag

  const int tid  = threadIdx.x;
  const int lane = tid & 31;
  const int wid  = tid >> 5;
  const int q0   = blockIdx.x * 128;
  const int b    = blockIdx.y / H_;
  const int h    = blockIdx.y % H_;

  const size_t rs = 3 * D_;
  const float* qbase = qkv + (size_t)(b * S_) * rs + h * DH_;
  const float* kbase = qbase + D_;
  const float* vbase = qbase + 2 * D_;

  // Q block 128x64, scaled, into A-fragment layout
  #pragma unroll
  for (int i = 0; i < 16; ++i) {
    int p = tid + i * 256;              // 4096 dh-pairs
    int row = p >> 5;
    int jp  = p & 31;                   // dh = 2*jp
    const float* src = qbase + (size_t)(q0 + row) * rs + 2 * jp;
    unsigned pk = pack_bf16(src[0] * 0.125f, src[1] * 0.125f);
    int kf  = jp >> 4;
    int jpp = jp & 15;
    int tl  = (row & 15) + 16 * ((jpp >> 2) & 1);
    int ep  = (jpp & 3) + 4 * (jpp >> 3);
    q_lds[((row >> 4) * 2 + kf) * 256 + tl * 8 + ep] = pk;
  }
  __syncthreads();

  v16bf qf0 = *(const v16bf*)&q_lds[(wid * 2 + 0) * 256 + lane * 8];
  v16bf qf1 = *(const v16bf*)&q_lds[(wid * 2 + 1) * 256 + lane * 8];

  v8f vz = {};
  v8f acc_o[4];
  float m_i[8], l_i[8];
  #pragma unroll
  for (int v = 0; v < 8; ++v) { m_i[v] = -1e30f; l_i[v] = 0.0f; }
  #pragma unroll
  for (int nt = 0; nt < 4; ++nt) acc_o[nt] = vz;

  const int nlo = lane & 15;
  const int hh  = lane >> 4;
  const int jmax = CAUSAL ? (q0 + 128) : S_;

  for (int j0 = 0; j0 < jmax; j0 += 32) {
    __syncthreads();
    // K chunk 32x64 -> B frags (contraction = dh, N = key)
    #pragma unroll
    for (int i = 0; i < 4; ++i) {
      int p = tid + i * 256;            // 1024 dh-pairs
      int row = p >> 5;                 // key-in-chunk
      int jp  = p & 31;
      const float* src = kbase + (size_t)(j0 + row) * rs + 2 * jp;
      unsigned pk = pack_bf16(src[0], src[1]);
      int kf  = jp >> 4;
      int jpp = jp & 15;
      int tl  = (row & 15) + 16 * ((jpp >> 2) & 1);
      int ep  = (jpp & 3) + 4 * (jpp >> 3);
      k_lds[((row >> 4) * 2 + kf) * 256 + tl * 8 + ep] = pk;
    }
    // V chunk 32x64 -> B frags (contraction = key, N = dh)
    #pragma unroll
    for (int i = 0; i < 8; ++i) {
      int p = tid + i * 256;            // 2048 scalars
      int key = p >> 6;
      int dh  = p & 63;
      float f = vbase[(size_t)(j0 + key) * rs + dh];
      int tl = (dh & 15) + 16 * ((key >> 3) & 1);
      v_lds[(dh >> 4) * 512 + tl * 16 + (key & 7) + 8 * (key >> 4)] = bf16_bits(f);
    }
    __syncthreads();

    // scores 16x32 for this wave's queries
    v8f s[2]; s[0] = vz; s[1] = vz;
    #pragma unroll
    for (int nt = 0; nt < 2; ++nt) {
      v16bf kf0 = *(const v16bf*)&k_lds[(nt * 2 + 0) * 256 + lane * 8];
      v16bf kf1 = *(const v16bf*)&k_lds[(nt * 2 + 1) * 256 + lane * 8];
      s[nt] = __builtin_amdgcn_wmma_f32_16x16x32_bf16(false, qf0, false, kf0, (short)0, s[nt], false, false);
      s[nt] = __builtin_amdgcn_wmma_f32_16x16x32_bf16(false, qf1, false, kf1, (short)0, s[nt], false, false);
    }
    if (CAUSAL) {
      #pragma unroll
      for (int nt = 0; nt < 2; ++nt) {
        int key = j0 + nt * 16 + nlo;
        #pragma unroll
        for (int v = 0; v < 8; ++v) {
          int qrow = q0 + wid * 16 + v + 8 * hh;
          if (key > qrow) s[nt][v] = -1e30f;
        }
      }
    }

    float mnew[8], alpha[8];
    #pragma unroll
    for (int v = 0; v < 8; ++v) {
      float m = fmaxf(s[0][v], s[1][v]);
      m = fmaxf(m, __shfl_xor(m, 1, 32));
      m = fmaxf(m, __shfl_xor(m, 2, 32));
      m = fmaxf(m, __shfl_xor(m, 4, 32));
      m = fmaxf(m, __shfl_xor(m, 8, 32));
      mnew[v]  = fmaxf(m_i[v], m);
      alpha[v] = __expf(m_i[v] - mnew[v]);
      m_i[v]   = mnew[v];
    }
    float pv[2][8];
    #pragma unroll
    for (int nt = 0; nt < 2; ++nt)
      #pragma unroll
      for (int v = 0; v < 8; ++v)
        pv[nt][v] = __expf(s[nt][v] - mnew[v]);
    #pragma unroll
    for (int v = 0; v < 8; ++v) {
      float su = pv[0][v] + pv[1][v];
      su += __shfl_xor(su, 1, 32);
      su += __shfl_xor(su, 2, 32);
      su += __shfl_xor(su, 4, 32);
      su += __shfl_xor(su, 8, 32);
      l_i[v] = l_i[v] * alpha[v] + su;
    }
    #pragma unroll
    for (int nt = 0; nt < 4; ++nt)
      #pragma unroll
      for (int v = 0; v < 8; ++v)
        acc_o[nt][v] *= alpha[v];

    // transpose P (C layout) -> per-wave A-fragment layout via LDS
    #pragma unroll
    for (int nt = 0; nt < 2; ++nt)
      #pragma unroll
      for (int v = 0; v < 8; ++v) {
        int c  = nt * 16 + nlo;
        int r  = v + 8 * hh;
        int tl = r + 16 * ((c >> 3) & 1);
        int te = (c & 7) + 8 * ((c >> 4) & 1);
        p_lds[wid * 512 + tl * 16 + te] = bf16_bits(pv[nt][v]);
      }
    v16bf pf = *(const v16bf*)&p_lds[wid * 512 + lane * 16];
    #pragma unroll
    for (int nt = 0; nt < 4; ++nt) {
      v16bf vf = *(const v16bf*)&v_lds[nt * 512 + lane * 16];
      acc_o[nt] = __builtin_amdgcn_wmma_f32_16x16x32_bf16(
          false, pf, false, vf, (short)0, acc_o[nt], false, false);
    }
  }

  float* obase = ctx + (size_t)(b * S_) * D_ + h * DH_;
  #pragma unroll
  for (int nt = 0; nt < 4; ++nt)
    #pragma unroll
    for (int v = 0; v < 8; ++v) {
      int q  = q0 + wid * 16 + v + 8 * hh;
      int dh = nt * 16 + nlo;
      obase[(size_t)q * D_ + dh] = acc_o[nt][v] / l_i[v];
    }
}

// ---------------------------------------------------------------------------
// xout = LayerNorm(xin + add) * w + b ; one wave (32 lanes) per 768-wide row
// ---------------------------------------------------------------------------
__global__ __launch_bounds__(256) void ln_kernel(
    const float* __restrict__ xin, const float* __restrict__ add,
    const float* __restrict__ w, const float* __restrict__ bp,
    float* __restrict__ xout)
{
  const int lane = threadIdx.x & 31;
  const int wid  = threadIdx.x >> 5;
  const size_t row = (size_t)blockIdx.x * 8 + wid;
  const float* xr = xin + row * D_;
  const float* ar = add + row * D_;
  float vals[24];
  float s = 0.0f;
  #pragma unroll
  for (int j = 0; j < 24; ++j) {
    vals[j] = xr[lane + 32 * j] + ar[lane + 32 * j];
    s += vals[j];
  }
  s += __shfl_xor(s, 1, 32); s += __shfl_xor(s, 2, 32);
  s += __shfl_xor(s, 4, 32); s += __shfl_xor(s, 8, 32);
  s += __shfl_xor(s, 16, 32);
  float mu = s * (1.0f / (float)D_);
  float vv = 0.0f;
  #pragma unroll
  for (int j = 0; j < 24; ++j) { float d = vals[j] - mu; vv += d * d; }
  vv += __shfl_xor(vv, 1, 32); vv += __shfl_xor(vv, 2, 32);
  vv += __shfl_xor(vv, 4, 32); vv += __shfl_xor(vv, 8, 32);
  vv += __shfl_xor(vv, 16, 32);
  float rstd = rsqrtf(vv * (1.0f / (float)D_) + 1e-5f);
  #pragma unroll
  for (int j = 0; j < 24; ++j) {
    int e = lane + 32 * j;
    xout[row * D_ + e] = (vals[j] - mu) * rstd * w[e] + bp[e];
  }
}

// ---------------------------------------------------------------------------
// patches [B, S, OUT] -> out [B, 3, IMG, IMG]
// ---------------------------------------------------------------------------
__global__ __launch_bounds__(256) void permute_kernel(const float* __restrict__ pbuf,
                                                      float* __restrict__ out) {
  size_t idx = (size_t)blockIdx.x * 256 + threadIdx.x;   // B*3*IMG*IMG
  int j = (int)(idx % IMG_);
  int i = (int)((idx / IMG_) % IMG_);
  int c = (int)((idx / ((size_t)IMG_ * IMG_)) % 3);
  int b = (int)(idx / ((size_t)IMG_ * IMG_ * 3));
  out[idx] = pbuf[(size_t)b * (S_ * OUT_) + ((size_t)i * IMG_ + j) * 3 + c];
}

// ---------------------------------------------------------------------------
extern "C" void kernel_launch(void* const* d_in, const int* in_sizes, int n_in,
                              void* d_out, int out_size, void* d_ws, size_t ws_size,
                              hipStream_t stream) {
  (void)in_sizes; (void)n_in; (void)out_size; (void)ws_size;

  const float* enc = (const float*)d_in[0];
  const float* siw = (const float*)d_in[1];
  const float* sib = (const float*)d_in[2];
  const float* sow = (const float*)d_in[3];
  const float* sob = (const float*)d_in[4];
  const float* ciw = (const float*)d_in[5];
  const float* cib = (const float*)d_in[6];
  const float* cow = (const float*)d_in[7];
  const float* cob = (const float*)d_in[8];
  const float* lnw = (const float*)d_in[9];
  const float* lnb = (const float*)d_in[10];
  const float* f1w = (const float*)d_in[11];
  const float* f1b = (const float*)d_in[12];
  const float* f2w = (const float*)d_in[13];
  const float* f2b = (const float*)d_in[14];
  const float* tpw = (const float*)d_in[15];
  const float* tpb = (const float*)d_in[16];

  // workspace carve-up (all region sizes are multiples of 16 bytes)
  float*    x    = (float*)d_ws;                        // BS * D     fp32
  float*    qkvb = x    + (size_t)BS_ * D_;             // BS * 3D    fp32 (also FFN hidden)
  float*    ctx  = qkvb + (size_t)BS_ * 3 * D_;         // BS * D     fp32 (attn ctx / gemm out)
  unsigned* pA   = (unsigned*)(ctx + (size_t)BS_ * D_); // BS * DFF   bf16 packed activations
  unsigned* pEnc = pA  + (size_t)BS_ * DFF_ / 2;        // BS * D     bf16 packed enc (k/v source)
  unsigned* pW   = pEnc + (size_t)BS_ * D_ / 2;         // 3D * D     bf16 packed weights

  const dim3 blk(256);
  const int nPE = (int)(((size_t)BS_ * D_) / 256);      // 24576 blocks
  add_pe_kernel<<<nPE, blk, 0, stream>>>(enc, x);
  pack_a_kernel<<<BS_, blk, 0, stream>>>(enc, pEnc, D_);   // cross-attn k/v source, once

  const int KtD = D_ / 32;   // 24

  for (int l = 0; l < L_; ++l) {
    const float* siw_l = siw + (size_t)l * 3 * D_ * D_;
    const float* sib_l = sib + (size_t)l * 3 * D_;
    const float* sow_l = sow + (size_t)l * D_ * D_;
    const float* sob_l = sob + (size_t)l * D_;
    const float* ciw_l = ciw + (size_t)l * 3 * D_ * D_;
    const float* cib_l = cib + (size_t)l * 3 * D_;
    const float* cow_l = cow + (size_t)l * D_ * D_;
    const float* cob_l = cob + (size_t)l * D_;
    const float* f1w_l = f1w + (size_t)l * DFF_ * D_;
    const float* f1b_l = f1b + (size_t)l * DFF_;
    const float* f2w_l = f2w + (size_t)l * D_ * DFF_;
    const float* f2b_l = f2b + (size_t)l * D_;
    const float* lnw_l = lnw + (size_t)l * 3 * D_;
    const float* lnb_l = lnb + (size_t)l * 3 * D_;

    // --- self attention block ---
    pack_a_kernel<<<BS_, blk, 0, stream>>>(x, pA, D_);
    pack_b_kernel<<<3 * D_, blk, 0, stream>>>(siw_l, pW, D_);
    gemm_kernel<0><<<dim3(18, 64), blk, 0, stream>>>(pA, pW, sib_l, qkvb, D_, 3 * D_);
    attn_kernel<true><<<dim3(S_ / 128, B_ * H_), blk, 0, stream>>>(qkvb, ctx);
    pack_a_kernel<<<BS_, blk, 0, stream>>>(ctx, pA, D_);
    pack_b_kernel<<<D_, blk, 0, stream>>>(sow_l, pW, D_);
    gemm_kernel<0><<<dim3(6, 64), blk, 0, stream>>>(pA, pW, sob_l, ctx, D_, D_);
    ln_kernel<<<1024, blk, 0, stream>>>(x, ctx, lnw_l + 0 * D_, lnb_l + 0 * D_, x);

    // --- cross attention block (q from x, k/v from encoded_patches) ---
    pack_a_kernel<<<BS_, blk, 0, stream>>>(x, pA, D_);
    pack_b_kernel<<<3 * D_, blk, 0, stream>>>(ciw_l, pW, D_);
    gemm_kernel<0><<<dim3(6, 64), blk, 0, stream>>>(pA, pW, cib_l, qkvb, D_, 3 * D_);
    gemm_kernel<0><<<dim3(12, 64), blk, 0, stream>>>(
        pEnc, pW + (size_t)(D_ / 16) * KtD * 256, cib_l + D_, qkvb + D_, D_, 3 * D_);
    attn_kernel<false><<<dim3(S_ / 128, B_ * H_), blk, 0, stream>>>(qkvb, ctx);
    pack_a_kernel<<<BS_, blk, 0, stream>>>(ctx, pA, D_);
    pack_b_kernel<<<D_, blk, 0, stream>>>(cow_l, pW, D_);
    gemm_kernel<0><<<dim3(6, 64), blk, 0, stream>>>(pA, pW, cob_l, ctx, D_, D_);
    ln_kernel<<<1024, blk, 0, stream>>>(x, ctx, lnw_l + 1 * D_, lnb_l + 1 * D_, x);

    // --- FFN ---
    pack_a_kernel<<<BS_, blk, 0, stream>>>(x, pA, D_);
    pack_b_kernel<<<DFF_, blk, 0, stream>>>(f1w_l, pW, D_);
    gemm_kernel<1><<<dim3(16, 64), blk, 0, stream>>>(pA, pW, f1b_l, qkvb, D_, DFF_);
    pack_a_kernel<<<BS_, blk, 0, stream>>>(qkvb, pA, DFF_);
    pack_b_kernel<<<D_, blk, 0, stream>>>(f2w_l, pW, DFF_);
    gemm_kernel<0><<<dim3(6, 64), blk, 0, stream>>>(pA, pW, f2b_l, ctx, DFF_, D_);
    ln_kernel<<<1024, blk, 0, stream>>>(x, ctx, lnw_l + 2 * D_, lnb_l + 2 * D_, x);
  }

  // final projection + layout permute
  pack_a_kernel<<<BS_, blk, 0, stream>>>(x, pA, D_);
  pack_b_kernel<<<OUT_, blk, 0, stream>>>(tpw, pW, D_);
  gemm_kernel<0><<<dim3(6, 64), blk, 0, stream>>>(pA, pW, tpb, ctx, D_, OUT_);
  const int nPerm = (int)(((size_t)B_ * 3 * IMG_ * IMG_) / 256);
  permute_kernel<<<nPerm, blk, 0, stream>>>(ctx, (float*)d_out);
}
